// GGNN_29918742184218
// MI455X (gfx1250) — matrix operand order repaired and checked by reading, not compiled, not run on hardware
//
#include <hip/hip_runtime.h>

typedef __attribute__((ext_vector_type(16))) __bf16 v16bf;
typedef __attribute__((ext_vector_type(8)))  float  v8f;

#define BATCH 256
#define TLEN  40
#define UNIQ  40
#define DDIM  100
#define KDIM  200
#define KPAD  224          // 7 * 32
#define KSTEPS 7
#define NNODE 200000
#define NCOLS 199999       // N_NODE - 1
#define NB 4               // 16-col N tiles per wave
#define COLS_PER_WAVE 64
#define COLS_PER_BLOCK 512 // 8 waves * 64

// ---------------------------------------------------------------------------
// Kernel 1: attention prep -> ma[B,200] -> bf16 A[B,224] (zero padded K)
// One block per batch element, 256 threads (8 wave32).
// ---------------------------------------------------------------------------
__global__ __launch_bounds__(256) void ggnn_prep(
    const float* __restrict__ re_emb,    // [B][U][D]
    const float* __restrict__ re_embt,   // [B][U][D]
    const float* __restrict__ mask,      // [B][T]
    const float* __restrict__ mask_t,    // [B][T]
    const int*   __restrict__ alias_,    // [B][T]
    const int*   __restrict__ alias_t,   // [B][T]
    const float* __restrict__ w1,        // [200][200]
    const float* __restrict__ w2,        // [200][200]
    const float* __restrict__ w3,        // [200][200]
    const float* __restrict__ vvec,      // [200]
    const float* __restrict__ bvec,      // [200]
    __bf16*      __restrict__ Aout)      // [B][224]
{
    const int b    = blockIdx.x;
    const int tid  = threadIdx.x;
    const int lane = tid & 31;
    const int wave = tid >> 5;

    __shared__ float w2c[50 * 200];      // 40 KB chunk of w2
    __shared__ float lastvec[KDIM];
    __shared__ float lastW[KDIM];
    __shared__ float coefLds[TLEN];
    __shared__ float sLds[KDIM];
    __shared__ int   aliasLds[TLEN];
    __shared__ int   aliasTLds[TLEN];
    __shared__ int   lastIds[2];

    if (tid < TLEN) {
        aliasLds[tid]  = alias_[b * TLEN + tid];
        aliasTLds[tid] = alias_t[b * TLEN + tid];
    }
    if (tid == 0) {
        float s1 = 0.f, s2 = 0.f;
        for (int t = 0; t < TLEN; ++t) { s1 += mask[b*TLEN+t]; s2 += mask_t[b*TLEN+t]; }
        int rm  = (int)s1; if (rm  < 1) rm  = 1;
        int rmt = (int)s2; if (rmt < 1) rmt = 1;
        lastIds[0] = alias_[b*TLEN + rm  - 1];
        lastIds[1] = alias_t[b*TLEN + rmt - 1];
    }
    __syncthreads();

    if (tid < KDIM) {
        lastvec[tid] = (tid < DDIM)
            ? re_emb [(size_t)(b*UNIQ + lastIds[0]) * DDIM + tid]
            : re_embt[(size_t)(b*UNIQ + lastIds[1]) * DDIM + (tid - DDIM)];
    }
    __syncthreads();

    if (tid < KDIM) {
        float acc = bvec[tid];
        for (int k = 0; k < KDIM; ++k) acc = fmaf(lastvec[k], w1[k*KDIM + tid], acc);
        lastW[tid] = acc;
    }

    // seq = sh @ w2, t-blocked (5 t per wave), w2 streamed through LDS chunks
    float acc[5][7];
    #pragma unroll
    for (int tt = 0; tt < 5; ++tt)
        #pragma unroll
        for (int jj = 0; jj < 7; ++jj) acc[tt][jj] = 0.f;

    for (int kc = 0; kc < 4; ++kc) {
        __syncthreads();
        for (int i = tid; i < 50*200; i += 256) w2c[i] = w2[kc*50*200 + i];
        __syncthreads();

        const float* shp[5];
        #pragma unroll
        for (int tt = 0; tt < 5; ++tt) {
            int t = wave * 5 + tt;
            shp[tt] = (kc < 2)
                ? re_emb  + (size_t)(b*UNIQ + aliasLds[t])  * DDIM + kc*50
                : re_embt + (size_t)(b*UNIQ + aliasTLds[t]) * DDIM + (kc-2)*50;
        }
        for (int k = 0; k < 50; ++k) {
            float wv[7];
            #pragma unroll
            for (int jj = 0; jj < 7; ++jj) {
                int j = lane + 32*jj;
                wv[jj] = (j < KDIM) ? w2c[k*200 + j] : 0.f;
            }
            #pragma unroll
            for (int tt = 0; tt < 5; ++tt) {
                float sv = shp[tt][k];
                #pragma unroll
                for (int jj = 0; jj < 7; ++jj)
                    acc[tt][jj] = fmaf(sv, wv[jj], acc[tt][jj]);
            }
        }
    }

    // coef[t] = (sigmoid(lastW + seq) . v) * mask
    #pragma unroll
    for (int tt = 0; tt < 5; ++tt) {
        int t = wave * 5 + tt;
        float part = 0.f;
        #pragma unroll
        for (int jj = 0; jj < 7; ++jj) {
            int j = lane + 32*jj;
            if (j < KDIM) {
                float mval = 1.f / (1.f + __expf(-(lastW[j] + acc[tt][jj])));
                part = fmaf(mval, vvec[j], part);
            }
        }
        for (int o = 16; o > 0; o >>= 1) part += __shfl_xor(part, o, 32);
        if (lane == 0) coefLds[t] = part * mask[b*TLEN + t];
    }
    __syncthreads();

    // s = sum_t coef[t] * sh[t]   (re-gather, cheap)
    if (tid < KDIM) {
        float sv = 0.f;
        for (int t = 0; t < TLEN; ++t) {
            float sh = (tid < DDIM)
                ? re_emb [(size_t)(b*UNIQ + aliasLds[t])  * DDIM + tid]
                : re_embt[(size_t)(b*UNIQ + aliasTLds[t]) * DDIM + (tid - DDIM)];
            sv = fmaf(coefLds[t], sh, sv);
        }
        sLds[tid] = sv;
    }
    __syncthreads();

    // ma = s @ w3 ; emit bf16, zero-pad K to 224
    if (tid < KPAD) {
        float ma = 0.f;
        if (tid < KDIM)
            for (int k = 0; k < KDIM; ++k) ma = fmaf(sLds[k], w3[k*KDIM + tid], ma);
        Aout[(size_t)b * KPAD + tid] = (__bf16)ma;
    }
}

// ---------------------------------------------------------------------------
// Kernel 2: logits = ma @ b_embedding.T via v_wmma_f32_16x16x32_bf16.
// 256 threads = 8 waves; each wave owns 64 columns (NB=4 tiles), holds their
// B fragments in registers, streams A (256x224 bf16) through LDS in 2 phases.
// Streaming data (embedding reads, logits writes) uses non-temporal hints so
// the 205MB output stream does not evict the reused tables from the 192MB L2.
// ---------------------------------------------------------------------------
__global__ __launch_bounds__(256) void ggnn_logits_gemm(
    const __bf16* __restrict__ A,        // [256][224] bf16
    const float*  __restrict__ emb,      // [200000][100]
    const float*  __restrict__ embt,     // [1000][100]
    const int*    __restrict__ csort,    // [200000]
    float*        __restrict__ out)      // [256][199999]
{
    __shared__ __bf16 Alds[128 * KPAD];  // 57344 B per phase

    const int tid     = threadIdx.x;
    const int lane    = tid & 31;
    const int wave    = tid >> 5;
    const int halfsel = lane >> 4;       // 0: lanes 0-15, 1: lanes 16-31
    const int lane16  = lane & 15;

    const int colBase = blockIdx.x * COLS_PER_BLOCK + wave * COLS_PER_WAVE;

    // ---- Build B fragments (bf16) from fp32 gathers; contiguous K order used here (A matches):
    //      lanes 0-15 hold K = s*32 + h, lanes 16-31 hold K = s*32 + 16 + h.
    v16bf Bfrag[NB][KSTEPS];
    #pragma unroll
    for (int nb = 0; nb < NB; ++nb) {
        int col = colBase + nb*16 + lane16;
        int node = (col < NCOLS) ? (col + 1) : 0;
        const float* erow = emb  + (size_t)node * DDIM;
        const float* trow = embt + (size_t)csort[node] * DDIM;
        #pragma unroll
        for (int s = 0; s < KSTEPS; ++s) {
            int kbase = s*32 + (halfsel ? 16 : 0);
            v16bf bf;
            #pragma unroll
            for (int h = 0; h < 16; ++h) {
                int k = kbase + h;
                float f = 0.f;
                if (k < DDIM)      f = __builtin_nontemporal_load(erow + k); // single-use stream
                else if (k < KDIM) f = trow[k - DDIM];                       // L2-hot table
                bf[h] = (__bf16)f;
            }
            Bfrag[nb][s] = bf;
        }
    }

    const int aoff = halfsel * 16;       // A layout: lanes>=16 shifted by 16 in K
    const v8f zero = {0.f,0.f,0.f,0.f,0.f,0.f,0.f,0.f};

    #pragma unroll 1
    for (int phase = 0; phase < 2; ++phase) {
        __syncthreads();
        {   // cooperative copy of 128 A rows into LDS (ds traffic amortized)
            const uint4* src = (const uint4*)(A + (size_t)phase * 128 * KPAD);
            uint4* dst = (uint4*)Alds;
            for (int i = tid; i < 3584; i += 256) dst[i] = src[i];
        }
        __syncthreads();

        #pragma unroll 1
        for (int m = 0; m < 8; ++m) {
            const __bf16* arow = Alds + (m*16 + lane16) * KPAD;
            v8f c[NB];
            #pragma unroll
            for (int nb = 0; nb < NB; ++nb) c[nb] = zero;

            #pragma unroll
            for (int s = 0; s < KSTEPS; ++s) {
                // A fragment: two ds_load_b128, contiguous K order matching B
                union { v16bf v; struct { uint4 lo; uint4 hi; } u; } afu;
                afu.u.lo = *(const uint4*)(arow + s*32 + aoff);
                afu.u.hi = *(const uint4*)(arow + s*32 + 8 + aoff);
                v16bf af = afu.v;
                #pragma unroll
                for (int nb = 0; nb < NB; ++nb)
                    c[nb] = __builtin_amdgcn_wmma_f32_16x16x32_bf16(
                        false, af, false, Bfrag[nb][s],
                        (short)0, c[nb], false, false);
            }

            // store C: VGPR r -> row m*16 + r + 8*halfsel, col = lane16
            const int globRow = phase * 128 + m * 16 + halfsel * 8;
            #pragma unroll
            for (int nb = 0; nb < NB; ++nb) {
                int col = colBase + nb*16 + lane16;
                if (col < NCOLS) {
                    float* op = out + (size_t)globRow * NCOLS + col;
                    #pragma unroll
                    for (int r = 0; r < 8; ++r)
                        __builtin_nontemporal_store(c[nb][r], op + (size_t)r * NCOLS);
                }
            }
        }
    }
}

// ---------------------------------------------------------------------------
extern "C" void kernel_launch(void* const* d_in, const int* in_sizes, int n_in,
                              void* d_out, int out_size, void* d_ws, size_t ws_size,
                              hipStream_t stream) {
    const float* re_emb  = (const float*)d_in[0];
    const float* re_embt = (const float*)d_in[1];
    const float* mask    = (const float*)d_in[2];
    const float* mask_t  = (const float*)d_in[3];
    const int*   alias_  = (const int*)  d_in[4];
    const int*   alias_t = (const int*)  d_in[5];
    const int*   csort   = (const int*)  d_in[6];
    const float* emb     = (const float*)d_in[7];
    const float* embt    = (const float*)d_in[8];
    const float* w1      = (const float*)d_in[9];
    const float* w2      = (const float*)d_in[10];
    const float* w3      = (const float*)d_in[11];
    const float* vvec    = (const float*)d_in[12];
    const float* bvec    = (const float*)d_in[13];
    float*       out     = (float*)d_out;

    __bf16* Abf = (__bf16*)d_ws;   // 256*224*2 = 114,688 bytes of scratch

    ggnn_prep<<<BATCH, 256, 0, stream>>>(re_emb, re_embt, mask, mask_t,
                                         alias_, alias_t, w1, w2, w3,
                                         vvec, bvec, Abf);

    const int grid = (NCOLS + COLS_PER_BLOCK - 1) / COLS_PER_BLOCK; // 391
    ggnn_logits_gemm<<<grid, 256, 0, stream>>>(Abf, emb, embt, csort, out);
}
